// DeformConv2d_23845658427899
// MI455X (gfx1250) — matrix operand order repaired and checked by reading, not compiled, not run on hardware
//
#include <hip/hip_runtime.h>

typedef __attribute__((ext_vector_type(16))) _Float16 v16h;
typedef __attribute__((ext_vector_type(8)))  _Float16 v8h;
typedef __attribute__((ext_vector_type(8)))  float    v8f;
typedef __attribute__((ext_vector_type(4)))  unsigned int v4u;
typedef __attribute__((ext_vector_type(8)))  int      v8i;
typedef __attribute__((ext_vector_type(4)))  int      v4i;

#define B_   8
#define CIN  256
#define H_   64
#define W_   64
#define COUT 256
#define KH   3
#define KW   3
#define HO   64
#define WO   64
#define P_   (HO * WO)          // 4096
#define TAPS (KH * KW)          // 9
#define CK   32                 // K-chunk per WMMA (channels)
#define NCHUNK (TAPS * (CIN / CK))   // 72 K-chunks total
#define BP   32                 // pixels per block tile (M)
#define BCO  256                // couts per block tile (N) -- full Cout
#define LDSPAD 8
#define AS_STRIDE (CK + LDSPAD) // 40 halfs = 80 B rows (16B aligned, bank-friendly)
#define BS_STRIDE (CK + LDSPAD)

// ---------------------------------------------------------------------------
// Repack weight (Cout, Cin, 3, 3) f32 -> wrep[tap][cout][cin] f16 (contig in cin)
__global__ void repack_weight(const float* __restrict__ w, _Float16* __restrict__ wrep) {
    int idx = blockIdx.x * blockDim.x + threadIdx.x;   // < TAPS*COUT*CIN = 589824
    int k   = idx >> 16;                               // COUT*CIN == 65536 exactly
    int rem = idx & 0xFFFF;
    int co  = rem >> 8;
    int c   = rem & 255;
    wrep[idx] = (_Float16)w[co * (CIN * TAPS) + c * TAPS + k];
}

// ---------------------------------------------------------------------------
// Tensor Data Mover: 2D tile (tile_dim0=32 ch contiguous, tile_dim1=256 couts,
// row stride 256 ch) from repacked f16 weights into LDS; pad 16 B every 64 B
// so rows land on the 80 B padded LDS pitch.
#if __has_builtin(__builtin_amdgcn_tensor_load_to_lds)
#define HAVE_TDM 1
__device__ __forceinline__ void tdm_load_B(unsigned lds_addr, const _Float16* gsrc) {
    unsigned long long ga = (unsigned long long)gsrc;
    v4u g0;
    g0.x = 1u;                                       // count=1, user descriptor
    g0.y = lds_addr;                                 // D# lds_addr [63:32]
    g0.z = (unsigned)(ga & 0xffffffffu);             // global_addr lo
    g0.w = (unsigned)((ga >> 32) & 0x01ffffffu)      // global_addr hi
         | (2u << 30);                               // type=2 ("image")
    v8i g1;
    g1[0] = (1 << 16)                                // data_size = 2 bytes
          | (1 << 20)                                // pad_enable
          | (3 << 22)                                // pad_interval: 16 DWORDs (64 B)
          | (3 << 25);                               // pad_amount: 4 DWORDs (16 B)
    g1[1] = (int)(256u << 16);                       // tensor_dim0 = 256 (lo16)
    g1[2] = (int)(256u << 16);                       // dim0 hi=0 | tensor_dim1 = 256
    g1[3] = (int)(32u << 16);                        // dim1 hi=0 | tile_dim0 = 32
    g1[4] = 256;                                     // tile_dim1 = 256, tile_dim2 = 0
    g1[5] = 256;                                     // tensor_dim0_stride = 256
    g1[6] = 0;
    g1[7] = 0;
    v4i gz = {0, 0, 0, 0};
#if __has_include(<hip/amd_detail/amd_gfx1250_TDM.h>)
    v8i g4 = {0, 0, 0, 0, 0, 0, 0, 0};               // therock clang-23: 6-arg form
    __builtin_amdgcn_tensor_load_to_lds(g0, g1, gz, gz, g4, 0);
#else
    __builtin_amdgcn_tensor_load_to_lds(g0, g1, gz, gz, 0);   // ROCm 7.2: 5-arg
#endif
}
#endif

__device__ __forceinline__ void stage_B(_Float16 (*bs)[BS_STRIDE], unsigned lds_addr,
                                        const _Float16* src, int t, int wv) {
#ifdef HAVE_TDM
    if (wv == 0) tdm_load_B(lds_addr, src);
#else
    const v8h* s = (const v8h*)(src + t * CIN);      // thread t -> cout row t
    *(v8h*)&bs[t][0]  = s[0]; *(v8h*)&bs[t][8]  = s[1];
    *(v8h*)&bs[t][16] = s[2]; *(v8h*)&bs[t][24] = s[3];
#endif
}
__device__ __forceinline__ void stage_B_wait(int wv) {
#ifdef HAVE_TDM
    if (wv == 0) __builtin_amdgcn_s_wait_tensorcnt(0);
#endif
}

// ---------------------------------------------------------------------------
struct Bilin { float w00, w01, w10, w11; int a00, a01, a10, a11; };

__device__ __forceinline__ Bilin bilin_setup(const float* offb, int pix,
                                             int oy, int ox, int k) {
    const int ti = k / KW, tj = k % KW;
    float offy = offb[(2 * k + 0) * P_ + pix];
    float offx = offb[(2 * k + 1) * P_ + pix];
    float py = (float)(oy - 1 + ti) + offy;
    float px = (float)(ox - 1 + tj) + offx;
    float y0f = floorf(py), x0f = floorf(px);
    int   y0 = (int)y0f,   x0 = (int)x0f;
    float ly = py - y0f,   lx = px - x0f;
    int   y1 = y0 + 1,     x1 = x0 + 1;
    Bilin bl;
    bl.w00 = (1.f - ly) * (1.f - lx); bl.w01 = (1.f - ly) * lx;
    bl.w10 = ly * (1.f - lx);         bl.w11 = ly * lx;
    if (y0 < 0 || y0 > H_ - 1) { bl.w00 = 0.f; bl.w01 = 0.f; }
    if (y1 < 0 || y1 > H_ - 1) { bl.w10 = 0.f; bl.w11 = 0.f; }
    if (x0 < 0 || x0 > W_ - 1) { bl.w00 = 0.f; bl.w10 = 0.f; }
    if (x1 < 0 || x1 > W_ - 1) { bl.w01 = 0.f; bl.w11 = 0.f; }
    int y0c = min(max(y0, 0), H_ - 1), y1c = min(max(y1, 0), H_ - 1);
    int x0c = min(max(x0, 0), W_ - 1), x1c = min(max(x1, 0), W_ - 1);
    bl.a00 = y0c * W_ + x0c; bl.a01 = y0c * W_ + x1c;
    bl.a10 = y1c * W_ + x0c; bl.a11 = y1c * W_ + x1c;
    return bl;
}

// ---------------------------------------------------------------------------
// One pipeline step: prefetch chunk IT+1 into buffers PP^1 (TDM for B, fused
// bilinear gather for A), run 4 WMMAs on buffers PP (chunk IT), then drain the
// DMA and barrier. PP is a compile-time constant (loop unrolled by 2).
#define PIPE_STEP(PP, IT)                                                     \
  {                                                                           \
    const int nxt = (IT) + 1;                                                 \
    if (nxt < NCHUNK) {                                                       \
      const int kn  = nxt >> 3;                                               \
      const int ccn = (nxt & 7) * CK;                                         \
      if ((nxt & 7) == 0) {                                                   \
        bl = bilin_setup(offb, pix, oy, ox, kn);                              \
        __builtin_prefetch(wrep + kn * (COUT * CIN) + t * CIN, 0, 1);         \
      }                                                                       \
      stage_B(Bs[(PP) ^ 1], bs_lds[(PP) ^ 1],                                 \
              wrep + kn * (COUT * CIN) + ccn, t, wv);                         \
      _Pragma("unroll")                                                       \
      for (int u = 0; u < 4; ++u) {                                           \
        const float* xc = xb + (ccn + cg * 4 + u) * P_;                       \
        float v = bl.w00 * xc[bl.a00] + bl.w01 * xc[bl.a01]                   \
                + bl.w10 * xc[bl.a10] + bl.w11 * xc[bl.a11];                  \
        As[(PP) ^ 1][lp][cg * 4 + u] = (_Float16)v;                           \
      }                                                                       \
    }                                                                         \
    {                                                                         \
      const _Float16* ap = &As[(PP)][mt * 16 + m][hf * 8];                    \
      v8h alo = *(const v8h*)ap;                                              \
      v8h ahi = *(const v8h*)(ap + 16);                                       \
      v16h afrag;                                                             \
      _Pragma("unroll")                                                       \
      for (int e = 0; e < 8; ++e) { afrag[e] = alo[e]; afrag[8+e] = ahi[e]; } \
      _Pragma("unroll")                                                       \
      for (int j = 0; j < 4; ++j) {                                           \
        const _Float16* bp = &Bs[(PP)][(nb + j) * 16 + m][hf * 8];            \
        v8h blv = *(const v8h*)bp;                                            \
        v8h bhv = *(const v8h*)(bp + 16);                                     \
        v16h bf;                                                              \
        _Pragma("unroll")                                                     \
        for (int e = 0; e < 8; ++e) { bf[e] = blv[e]; bf[8+e] = bhv[e]; }     \
        acc[j] = __builtin_amdgcn_wmma_f32_16x16x32_f16(                      \
            false, afrag, false, bf, (short)0, acc[j], false, false);         \
      }                                                                       \
    }                                                                         \
    if (nxt < NCHUNK) stage_B_wait(wv);                                       \
    __syncthreads();                                                          \
  }

// ---------------------------------------------------------------------------
__global__ __launch_bounds__(256)
void deform_conv_wmma(const float* __restrict__ x,
                      const float* __restrict__ offset,
                      const _Float16* __restrict__ wrep,
                      float* __restrict__ out)
{
    __shared__ _Float16 As[2][BP][AS_STRIDE];    // 2 x 32 pix  x 32 ch  ~5 KB
    __shared__ _Float16 Bs[2][BCO][BS_STRIDE];   // 2 x 256 cout x 32 ch ~40 KB

    const int t    = threadIdx.x;
    const int lane = t & 31;
    const int wv   = t >> 5;        // wave 0..7
    const int mt   = wv & 1;        // M tile (16 pix)
    const int nb   = (wv >> 1) * 4; // first of 4 N tiles owned by this wave
    const int m    = lane & 15;     // WMMA row/col within tile
    const int hf   = lane >> 4;     // lane half

    const int pix0 = blockIdx.x * BP;
    const int b    = blockIdx.y;

    // gather-phase mapping: 8 threads per pixel, 4 channels each
    const int lp  = t >> 3;
    const int cg  = t & 7;
    const int pix = pix0 + lp;
    const int oy  = pix >> 6;
    const int ox  = pix & 63;

    const float* xb   = x + b * (CIN * P_);
    const float* offb = offset + b * (2 * TAPS * P_);

    v8f acc[4] = {{}, {}, {}, {}};

    unsigned bs_lds[2];
    bs_lds[0] = (unsigned)(unsigned long long)(void*)&Bs[0][0][0];
    bs_lds[1] = (unsigned)(unsigned long long)(void*)&Bs[1][0][0];

    // ---- prologue: chunk 0 into buffers 0 ----
    Bilin bl = bilin_setup(offb, pix, oy, ox, 0);
    stage_B(Bs[0], bs_lds[0], wrep, t, wv);
    #pragma unroll
    for (int u = 0; u < 4; ++u) {
        const float* xc = xb + (cg * 4 + u) * P_;
        float v = bl.w00 * xc[bl.a00] + bl.w01 * xc[bl.a01]
                + bl.w10 * xc[bl.a10] + bl.w11 * xc[bl.a11];
        As[0][lp][cg * 4 + u] = (_Float16)v;
    }
    stage_B_wait(wv);
    __syncthreads();

    // ---- software-pipelined main loop, ping-pong unrolled by 2 ----
    for (int it = 0; it < NCHUNK; it += 2) {
        PIPE_STEP(0, it)
        PIPE_STEP(1, it + 1)
    }

    // ---- store D (f32): col N = lane%16 -> cout, row M = 8*hf + r -> pixel ----
    float* outb = out + b * (COUT * P_);
    #pragma unroll
    for (int j = 0; j < 4; ++j) {
        const int cb = (nb + j) * 16 + m;
        #pragma unroll
        for (int r = 0; r < 8; ++r) {
            int pixd = pix0 + mt * 16 + hf * 8 + r;
            outb[cb * P_ + pixd] = acc[j][r];
        }
    }
}

// ---------------------------------------------------------------------------
extern "C" void kernel_launch(void* const* d_in, const int* in_sizes, int n_in,
                              void* d_out, int out_size, void* d_ws, size_t ws_size,
                              hipStream_t stream) {
    const float* x      = (const float*)d_in[0];
    const float* offset = (const float*)d_in[1];
    const float* weight = (const float*)d_in[2];
    float* out = (float*)d_out;

    _Float16* wrep = (_Float16*)d_ws;   // 9*256*256 f16 = 1.125 MB

    repack_weight<<<(TAPS * COUT * CIN) / 256, 256, 0, stream>>>(weight, wrep);

    dim3 grid(P_ / BP, B_);             // (128, 8)
    deform_conv_wmma<<<grid, 256, 0, stream>>>(x, offset, wrep, out);
}
